// GAN_MPNN_86560770884157
// MI455X (gfx1250) — matrix operand order repaired and checked by reading, not compiled, not run on hardware
//
#include <hip/hip_runtime.h>

typedef __attribute__((ext_vector_type(16))) _Float16 v16h;
typedef __attribute__((ext_vector_type(8)))  float    v8f;

#define S_DIM 128

// Guaranteed native f32 global atomic add (no CAS fallback): VGLOBAL encoding,
// GV addressing (64-bit VGPR-pair address, SADDR off), no return -> STOREcnt.
__device__ __forceinline__ void atomic_add_f32(float* p, float v) {
    asm volatile("global_atomic_add_f32 %0, %1, off" :: "v"(p), "v"(v) : "memory");
}

__global__ __launch_bounds__(256) void fill_zero_f4(float4* p, long n4) {
    long i = (long)blockIdx.x * blockDim.x + threadIdx.x;
    if (i < n4) p[i] = make_float4(0.f, 0.f, 0.f, 0.f);
}

__global__ __launch_bounds__(256) void cvt_f32_to_f16(const float* __restrict__ src,
                                                      _Float16* __restrict__ dst, long n) {
    long i = (long)blockIdx.x * blockDim.x + threadIdx.x;
    if (i < n) dst[i] = (_Float16)src[i];
}

// Y[m][n] = relu( X[m][:] dot Wh[n][:] + bias[n] )      (accumulate==0)
// Y[m][n] += relu( ... )                                 (accumulate==1)
// One 16-row tile per block; wave w owns columns [16w,16w+16); K=128 = 4 WMMA ops.
__global__ __launch_bounds__(256)
void gemm_relu_wmma(const float* __restrict__ X, const _Float16* __restrict__ Wh,
                    const float* __restrict__ bias, float* __restrict__ Y,
                    int accumulate, int numRows) {
    const int lane   = threadIdx.x & 31;
    const int wave   = threadIdx.x >> 5;
    const int m0     = blockIdx.x * 16;
    const int n0     = wave * 16;
    const int hi     = lane >> 4;        // 0: lanes 0-15, 1: lanes 16-31
    const int l15    = lane & 15;

    int row = m0 + l15;
    if (row >= numRows) row = numRows - 1;   // keep EXEC all-ones for WMMA
    const float*    xrow = X  + (long)row * S_DIM;
    const int       ncol = n0 + l15;
    const _Float16* wrow = Wh + (long)ncol * S_DIM;

    v8f c = {0.f, 0.f, 0.f, 0.f, 0.f, 0.f, 0.f, 0.f};
#pragma unroll
    for (int kc = 0; kc < S_DIM; kc += 32) {
        // ---- A pack (16x32 f16): this lane holds row `row`,
        //      K = kc + 16*hi + {0..15} in a[0..15]
        const float* xa = xrow + kc + hi * 16;
        float4 q0 = *(const float4*)(xa + 0);
        float4 q1 = *(const float4*)(xa + 4);
        float4 q2 = *(const float4*)(xa + 8);
        float4 q3 = *(const float4*)(xa + 12);
        v16h a;
        a[0]  = (_Float16)q0.x; a[1]  = (_Float16)q0.y; a[2]  = (_Float16)q0.z; a[3]  = (_Float16)q0.w;
        a[4]  = (_Float16)q1.x; a[5]  = (_Float16)q1.y; a[6]  = (_Float16)q1.z; a[7]  = (_Float16)q1.w;
        a[8]  = (_Float16)q2.x; a[9]  = (_Float16)q2.y; a[10] = (_Float16)q2.z; a[11] = (_Float16)q2.w;
        a[12] = (_Float16)q3.x; a[13] = (_Float16)q3.y; a[14] = (_Float16)q3.z; a[15] = (_Float16)q3.w;
        // ---- B pack (32x16 f16): lane holds column ncol, K = kc + 16*hi + {0..15}
        //      B[k][n] = W[n][k] -> contiguous 32B load from W row ncol.
        v16h b = *(const v16h*)(wrow + kc + hi * 16);
        // D = A*B + C   (v_wmma_f32_16x16x32_f16)
        c = __builtin_amdgcn_wmma_f32_16x16x32_f16(false, a, false, b,
                                                   (short)0, c, false, false);
    }

    const float bv = bias[ncol];
#pragma unroll
    for (int v = 0; v < 8; ++v) {
        int m = m0 + v + hi * 8;            // C/D layout: VGPR v -> row v (+8 for hi lanes)
        if (m < numRows) {
            float val = c[v] + bv;
            val = val > 0.f ? val : 0.f;
            float* yp = Y + (long)m * S_DIM + ncol;
            if (accumulate) *yp = *yp + val; else *yp = val;
        }
    }
}

// One wave per edge: 32 lanes x float4 = 128 floats. agg[dst] += message[src].
__global__ __launch_bounds__(256)
void edge_scatter(const int* __restrict__ src, const int* __restrict__ dst,
                  const float* __restrict__ message, float* __restrict__ agg, long E) {
    long tid  = (long)blockIdx.x * blockDim.x + threadIdx.x;
    long e    = tid >> 5;
    int  lane = tid & 31;
    if (e >= E) return;
    int s = src[e];
    int d = dst[e];
    float4 v = *(const float4*)(message + (long)s * S_DIM + lane * 4);
    float* a = agg + (long)d * S_DIM + lane * 4;
    atomic_add_f32(a + 0, v.x);
    atomic_add_f32(a + 1, v.y);
    atomic_add_f32(a + 2, v.z);
    atomic_add_f32(a + 3, v.w);
}

// One wave per node: out[batch[node]] += state[node].
__global__ __launch_bounds__(256)
void segment_sum(const int* __restrict__ batch, const float* __restrict__ state,
                 float* __restrict__ out, int N) {
    long tid  = (long)blockIdx.x * blockDim.x + threadIdx.x;
    long node = tid >> 5;
    int  lane = tid & 31;
    if (node >= N) return;
    int g = batch[node];
    float4 v = *(const float4*)(state + node * S_DIM + lane * 4);
    float* o = out + (long)g * S_DIM + lane * 4;
    atomic_add_f32(o + 0, v.x);
    atomic_add_f32(o + 1, v.y);
    atomic_add_f32(o + 2, v.z);
    atomic_add_f32(o + 3, v.w);
}

extern "C" void kernel_launch(void* const* d_in, const int* in_sizes, int n_in,
                              void* d_out, int out_size, void* d_ws, size_t ws_size,
                              hipStream_t stream) {
    (void)n_in; (void)ws_size;
    // inputs: 0:x (unused), 1:W_msg[R,S,S], 2:b_msg[R,S], 3:W_upd[R,S,S], 4:b_upd[R,S],
    //         5:edge_index[2,E], 6:batch[N]
    const float* W_msg = (const float*)d_in[1];
    const float* b_msg = (const float*)d_in[2];
    const float* W_upd = (const float*)d_in[3];
    const float* b_upd = (const float*)d_in[4];
    const int*   eidx  = (const int*)d_in[5];
    const int*   batch = (const int*)d_in[6];

    const int  N = in_sizes[6];
    const long E = (long)in_sizes[5] / 2;
    const int  R = in_sizes[2] / S_DIM;
    float* out = (float*)d_out;

    // workspace carve-up (all offsets 256B-aligned: N*S*4 is a multiple of 512)
    float* state   = (float*)d_ws;
    float* message = state   + (long)N * S_DIM;
    float* agg     = message + (long)N * S_DIM;
    _Float16* WhM  = (_Float16*)(agg + (long)N * S_DIM);
    _Float16* WhU  = WhM + (long)R * S_DIM * S_DIM;

    const long nw = (long)R * S_DIM * S_DIM;
    cvt_f32_to_f16<<<(int)((nw + 255) / 256), 256, 0, stream>>>(W_msg, WhM, nw);
    cvt_f32_to_f16<<<(int)((nw + 255) / 256), 256, 0, stream>>>(W_upd, WhU, nw);

    const long n4 = (long)N * S_DIM / 4;
    fill_zero_f4<<<(int)((n4 + 255) / 256), 256, 0, stream>>>((float4*)state, n4);

    const int gemmBlocks = (N + 15) / 16;
    const long eThreads  = E * 32;
    for (int r = 0; r < R; ++r) {
        gemm_relu_wmma<<<gemmBlocks, 256, 0, stream>>>(
            state, WhM + (long)r * S_DIM * S_DIM, b_msg + r * S_DIM, message, 0, N);
        fill_zero_f4<<<(int)((n4 + 255) / 256), 256, 0, stream>>>((float4*)agg, n4);
        edge_scatter<<<(int)((eThreads + 255) / 256), 256, 0, stream>>>(
            eidx, eidx + E, message, agg, E);
        gemm_relu_wmma<<<gemmBlocks, 256, 0, stream>>>(
            agg, WhU + (long)r * S_DIM * S_DIM, b_upd + r * S_DIM, state, 1, N);
    }

    const long o4 = (long)out_size / 4;
    fill_zero_f4<<<(int)((o4 + 255) / 256), 256, 0, stream>>>((float4*)out, o4);
    const long nThreads = (long)N * 32;
    segment_sum<<<(int)((nThreads + 255) / 256), 256, 0, stream>>>(batch, state, out, N);
}
